// ExLoss_74483322847821
// MI455X (gfx1250) — matrix operand order repaired and checked by reading, not compiled, not run on hardware
//
#include <hip/hip_runtime.h>
#include <hip/hip_bf16.h>

// ---------------------------------------------------------------------------
// ExLoss for MI455X (gfx1250): bf16 WMMA GEMM for outputs = inputs @ V.T,
// wave-per-dot-product kernels for the hard-negative loss.
//
// GEMM register blocking: each wave owns a 64(M) x 64(N) tile = 4x4 WMMA
// accumulators, so each k-step's 8 fragment loads/converts feed 16 WMMAs
// (~4 issue slots per WMMA) -> kernel sits on the HBM roofline (~146 MB
// traffic @ 23.3 TB/s) instead of being VALU/issue bound.
// ---------------------------------------------------------------------------

typedef __attribute__((ext_vector_type(16))) __bf16 v16bf;
typedef __attribute__((ext_vector_type(8)))  float  v8f;

__device__ __forceinline__ void cvt8(const float4 lo, const float4 hi, v16bf& f, int base) {
    f[base + 0] = (__bf16)lo.x; f[base + 1] = (__bf16)lo.y;
    f[base + 2] = (__bf16)lo.z; f[base + 3] = (__bf16)lo.w;
    f[base + 4] = (__bf16)hi.x; f[base + 5] = (__bf16)hi.y;
    f[base + 6] = (__bf16)hi.z; f[base + 7] = (__bf16)hi.w;
}

// B fragment: 16 contiguous fp32 of one V row -> bf16, elements 0..15 = K order.
__device__ __forceinline__ v16bf load_cvtB(const float* __restrict__ p) {
    v16bf f;
    cvt8(*(const float4*)(p + 0), *(const float4*)(p + 4), f, 0);
    cvt8(*(const float4*)(p + 8), *(const float4*)(p + 12), f, 8);
    return f;
}

// A fragment: 16 contiguous fp32 of one A row -> bf16, elements 0..15 = K order.
__device__ __forceinline__ v16bf load_cvtA(const float* __restrict__ p) {
    v16bf f;
    cvt8(*(const float4*)(p +  0), *(const float4*)(p +  4), f, 0);
    cvt8(*(const float4*)(p +  8), *(const float4*)(p + 12), f, 8);
    return f;
}

// ---------------------------------------------------------------------------
// GEMM: Out[M,N] = A[M,K] @ Vm[N,K]^T   (fp32 in, fp32 out, bf16 WMMA inside)
// Block: 256 threads = 8 waves laid out 4(M) x 2(N).
// Block tile: 256(M) x 128(N). Wave tile: 64(M) x 64(N) = 4x4 WMMA accs.
// ---------------------------------------------------------------------------
__global__ __launch_bounds__(256)
void exloss_gemm_wmma(const float* __restrict__ A,
                      const float* __restrict__ Vm,
                      float* __restrict__ Out,
                      int M, int N, int K)
{
    const int lane = threadIdx.x & 31;
    const int wave = threadIdx.x >> 5;
    const int l15  = lane & 15;
    const int hi   = lane >> 4;            // 0 or 1 (K-half selector)

    const int mBase = blockIdx.y * 256 + (wave & 3) * 64;
    const int nBase = blockIdx.x * 128 + (wave >> 2) * 64;

    v8f acc[4][4];
    #pragma unroll
    for (int i = 0; i < 4; ++i)
        #pragma unroll
        for (int j = 0; j < 4; ++j)
            acc[i][j] = (v8f){};

    // Per-lane base pointers (contiguous k = 16*hi + s in A and B; both operands K-contiguous).
    const float* arow[4];
    const float* brow[4];
    #pragma unroll
    for (int i = 0; i < 4; ++i)
        arow[i] = A  + (size_t)(mBase + 16 * i + l15) * K + hi * 16;
    #pragma unroll
    for (int j = 0; j < 4; ++j)
        brow[j] = Vm + (size_t)(nBase + 16 * j + l15) * K + hi * 16;

    for (int k0 = 0; k0 < K; k0 += 32) {
        v16bf af[4], bf[4];
        #pragma unroll
        for (int i = 0; i < 4; ++i) af[i] = load_cvtA(arow[i] + k0);
        #pragma unroll
        for (int j = 0; j < 4; ++j) bf[j] = load_cvtB(brow[j] + k0);

        #pragma unroll
        for (int i = 0; i < 4; ++i)
            #pragma unroll
            for (int j = 0; j < 4; ++j)
                acc[i][j] = __builtin_amdgcn_wmma_f32_16x16x32_bf16(
                    false, af[i], false, bf[j], (short)0, acc[i][j], false, false);
    }

    // C/D layout: lane (0-15 | 16-31) = N l15, VGPR r = row r / r+8.
    #pragma unroll
    for (int i = 0; i < 4; ++i) {
        const int mRow = mBase + 16 * i + hi * 8;
        #pragma unroll
        for (int r = 0; r < 8; ++r) {
            float* orow = Out + (size_t)(mRow + r) * N + nBase + l15;
            #pragma unroll
            for (int j = 0; j < 4; ++j)
                orow[16 * j] = acc[i][j][r];
        }
    }
}

// ---------------------------------------------------------------------------
// Gate: any row of V with sum == 0 ?  (one wave per row; V is L2-resident)
// ---------------------------------------------------------------------------
__global__ __launch_bounds__(256)
void exloss_rowcheck(const float* __restrict__ Vm, int N, int K, int* __restrict__ hasZero)
{
    const int wid  = (int)((blockIdx.x * blockDim.x + threadIdx.x) >> 5);
    const int lane = threadIdx.x & 31;
    if (wid >= N) return;
    const float* row = Vm + (size_t)wid * K;
    float s = 0.0f;
    for (int k = lane; k < K; k += 32) s += row[k];
    for (int m = 16; m; m >>= 1) s += __shfl_xor(s, m, 32);
    if (lane == 0 && s == 0.0f) atomicOr(hasZero, 1);
}

// ---------------------------------------------------------------------------
// n_thrds[b] = dot(x_b, V[t_b]) / max(||x_b||, eps) - 0.3   (one wave per b)
// ---------------------------------------------------------------------------
__global__ __launch_bounds__(256)
void exloss_nthrd(const float* __restrict__ X, const int* __restrict__ tgt,
                  const float* __restrict__ Vm, int Bn, int K,
                  float* __restrict__ nthrd)
{
    const int wid  = (int)((blockIdx.x * blockDim.x + threadIdx.x) >> 5);
    const int lane = threadIdx.x & 31;
    if (wid >= Bn) return;
    const float* xr = X  + (size_t)wid * K;
    const float* vr = Vm + (size_t)tgt[wid] * K;
    float d = 0.0f, n2 = 0.0f;
    for (int k = lane; k < K; k += 32) {
        const float x = xr[k];
        d  = fmaf(x, vr[k], d);
        n2 = fmaf(x, x, n2);
    }
    for (int m = 16; m; m >>= 1) { d += __shfl_xor(d, m, 32); n2 += __shfl_xor(n2, m, 32); }
    if (lane == 0)
        nthrd[wid] = d / fmaxf(sqrtf(n2), 1e-12f) - 0.3f;
}

// ---------------------------------------------------------------------------
// Hard-negative pass: nsims[b,n] = dot(V[t_b], V[cid]); masked softplus sum.
// One wave per (b, n) pair. V rows are L2 hits (V fits in 192 MB L2).
// ---------------------------------------------------------------------------
__global__ __launch_bounds__(256)
void exloss_hn(const float* __restrict__ Vm, const int* __restrict__ tgt,
               const int* __restrict__ pairs, const int* __restrict__ l2c,
               const float* __restrict__ nthrd, int Bn, int NP, int K,
               float* __restrict__ sumAcc, int* __restrict__ cntAcc)
{
    const int wid  = (int)((blockIdx.x * blockDim.x + threadIdx.x) >> 5);
    const int lane = threadIdx.x & 31;
    if (wid >= Bn * NP) return;
    const int b = wid / NP;
    const int n = wid - b * NP;
    const int cid = l2c[pairs[b * NP + n]];
    const float* tr = Vm + (size_t)tgt[b] * K;
    const float* nr = Vm + (size_t)cid * K;
    float d = 0.0f;
    for (int k = lane; k < K; k += 32) d = fmaf(tr[k], nr[k], d);
    for (int m = 16; m; m >>= 1) d += __shfl_xor(d, m, 32);
    if (lane == 0) {
        if (d > nthrd[b] && d < 0.999999f) {
            // numerically-stable softplus
            const float sp = (d > 0.0f) ? d + log1pf(expf(-d)) : log1pf(expf(d));
            atomicAdd(sumAcc, sp);
            atomicAdd(cntAcc, 1);
        }
    }
}

__global__ void exloss_init(float* sumAcc, int* cntAcc, int* hasZero)
{
    *sumAcc = 0.0f; *cntAcc = 0; *hasZero = 0;
}

__global__ void exloss_final(const float* __restrict__ sumAcc, const int* __restrict__ cntAcc,
                             const int* __restrict__ hasZero, float* __restrict__ out)
{
    const float cnt = (float)(*cntAcc);
    float l = (cnt > 0.0f) ? (*sumAcc / fmaxf(cnt, 1.0f)) : 0.0f;
    if (*hasZero) l = 0.0f;
    out[0] = l;   // w_th = 1.0; w_bu = w_h = 0
}

// ---------------------------------------------------------------------------
extern "C" void kernel_launch(void* const* d_in, const int* in_sizes, int n_in,
                              void* d_out, int out_size, void* d_ws, size_t ws_size,
                              hipStream_t stream)
{
    const float* X     = (const float*)d_in[0];  // inputs  [B, D]
    const int*   tgt   = (const int*)  d_in[1];  // targets [B]
    const int*   pairs = (const int*)  d_in[2];  // neg_pairs [B, NP]
    /* d_in[3] = indexs (unused, w_h = 0) */
    const int*   l2c   = (const int*)  d_in[4];  // all_label_to_clusterid [C]
    const float* Vm    = (const float*)d_in[5];  // V [C, D]

    const int Bn = in_sizes[1];
    const int K  = in_sizes[0] / Bn;       // D = 2048
    const int C  = in_sizes[4];            // 16384
    const int NP = in_sizes[2] / Bn;       // 64

    float* out      = (float*)d_out;       // [loss, outputs...]
    float* outputs  = out + 1;             // [B, C] (dword aligned; scalar stores)

    float* ws_f   = (float*)d_ws;
    int*   ws_i   = (int*)d_ws;
    float* sumAcc = ws_f + 0;
    int*   cntAcc = ws_i + 1;
    int*   hz     = ws_i + 2;
    float* nthrd  = ws_f + 4;              // [B]

    exloss_init<<<dim3(1), dim3(1), 0, stream>>>(sumAcc, cntAcc, hz);

    // Big GEMM: grid = (N/128, M/256), 8 waves/block, 4x4 WMMA tiles per wave.
    dim3 gg(C / 128, Bn / 256);
    exloss_gemm_wmma<<<gg, dim3(256), 0, stream>>>(X, Vm, outputs, Bn, C, K);

    // One wave per V row (gate), per batch row (threshold), per neg pair (loss).
    exloss_rowcheck<<<dim3((C * 32 + 255) / 256), dim3(256), 0, stream>>>(Vm, C, K, hz);
    exloss_nthrd  <<<dim3((Bn * 32 + 255) / 256), dim3(256), 0, stream>>>(X, tgt, Vm, Bn, K, nthrd);
    exloss_hn     <<<dim3((Bn * NP * 32 + 255) / 256), dim3(256), 0, stream>>>(
                      Vm, tgt, pairs, l2c, nthrd, Bn, NP, K, sumAcc, cntAcc);
    exloss_final  <<<dim3(1), dim3(1), 0, stream>>>(sumAcc, cntAcc, hz, out);
}
